// SwAV_46488726012483
// MI455X (gfx1250) — matrix operand order repaired and checked by reading, not compiled, not run on hardware
//
#include <hip/hip_runtime.h>

// ---------------------------------------------------------------------------
// SwAV loss on MI455X (gfx1250, wave32, WMMA)
// ---------------------------------------------------------------------------
typedef __attribute__((ext_vector_type(8)))  _Float16 h8;
typedef __attribute__((ext_vector_type(16))) _Float16 v16h;
typedef __attribute__((ext_vector_type(8)))  float    v8f;

constexpr int BS      = 1024;
constexpr int NCROPS  = 8;
constexpr int ROWS    = BS * NCROPS;   // 8192
constexpr int KDIM    = 256;
constexpr int NPROTO  = 3000;
constexpr int NPAD    = 3072;          // padded to multiple of 128
constexpr float INV_EPS = 20.0f;       // 1/0.05
constexpr float INV_T   = 10.0f;       // 1/0.1

// ------------------------- block reduction (wave32) -------------------------
__device__ __forceinline__ float blockReduceSum256(float v, float* sbuf) {
    #pragma unroll
    for (int o = 16; o > 0; o >>= 1) v += __shfl_xor(v, o, 32);
    const int lane = threadIdx.x & 31;
    const int w    = threadIdx.x >> 5;
    if (lane == 0) sbuf[w] = v;
    __syncthreads();
    if (w == 0) {
        v = (lane < 8) ? sbuf[lane] : 0.0f;
        #pragma unroll
        for (int o = 4; o > 0; o >>= 1) v += __shfl_xor(v, o, 32);
    }
    __syncthreads();
    return v;   // valid on thread 0
}

// ------------------------- misc small kernels -------------------------------
__global__ void zero_scalars(float* out, float* S) {
    if (threadIdx.x == 0) { out[0] = 0.0f; S[0] = 0.0f; S[1] = 0.0f; }
}

// Row-normalize src (rows x 256, f32) -> dst (f16). Rows >= srcRows -> zeros.
__global__ void rownorm_f16(const float* __restrict__ src,
                            _Float16* __restrict__ dst, int srcRows) {
    const int row = blockIdx.x;
    const int t   = threadIdx.x;          // 256 threads, one per column
    if (row >= srcRows) { dst[(size_t)row * KDIM + t] = (_Float16)0.0f; return; }
    __shared__ float sbuf[8];
    __shared__ float snorm;
    const float v  = src[(size_t)row * KDIM + t];
    const float ss = blockReduceSum256(v * v, sbuf);
    if (t == 0) snorm = rsqrtf(ss + 1e-30f);
    __syncthreads();
    dst[(size_t)row * KDIM + t] = (_Float16)(v * snorm);
}

// ------------------------- WMMA GEMM ----------------------------------------
// C[ROWS x NPAD] = A[ROWS x 256] * B[NPAD x 256]^T, f16 in / f32 out.
// 256 threads = 8 waves; wave grid 4(M) x 2(N); each wave 32x64 (2x4 tiles).
constexpr int LDS_STRIDE_H = 40;          // 32 data halves + 8 pad = 80 bytes

__global__ __launch_bounds__(256)
void gemm_f16_wmma(const _Float16* __restrict__ A, const _Float16* __restrict__ B,
                   float* __restrict__ C) {
    __shared__ __align__(16) _Float16 sA[128 * LDS_STRIDE_H];
    __shared__ __align__(16) _Float16 sB[128 * LDS_STRIDE_H];

    const int tid   = threadIdx.x;
    const int lane  = tid & 31;
    const int wave  = tid >> 5;
    const int wm    = wave & 3;           // 0..3 -> M offset wm*32
    const int wn    = wave >> 2;          // 0..1 -> N offset wn*64
    const int mBase = blockIdx.y * 128;
    const int nBase = blockIdx.x * 128;

    v8f acc[2][4];
    #pragma unroll
    for (int i = 0; i < 2; ++i)
        #pragma unroll
        for (int j = 0; j < 4; ++j)
            #pragma unroll
            for (int r = 0; r < 8; ++r) acc[i][j][r] = 0.0f;

    const int ldRow  = tid >> 1;          // 0..127
    const int ldHalf = tid & 1;           // 0..1 (16 halves = 32B each)
    const int hsel   = lane >> 4;         // 0/1 lane-half select
    const int lrow   = lane & 15;

    for (int k0 = 0; k0 < KDIM; k0 += 32) {
        __syncthreads();
        {   // stage A and B 128x32 f16 slabs into LDS (32B per thread each)
            const float4* sa = reinterpret_cast<const float4*>(
                A + (size_t)(mBase + ldRow) * KDIM + k0 + ldHalf * 16);
            float4* da = reinterpret_cast<float4*>(
                sA + ldRow * LDS_STRIDE_H + ldHalf * 16);
            da[0] = sa[0]; da[1] = sa[1];
            const float4* sb = reinterpret_cast<const float4*>(
                B + (size_t)(nBase + ldRow) * KDIM + k0 + ldHalf * 16);
            float4* db = reinterpret_cast<float4*>(
                sB + ldRow * LDS_STRIDE_H + ldHalf * 16);
            db[0] = sb[0]; db[1] = sb[1];
            if (k0 + 32 < KDIM) {         // global_prefetch_b8 next slab
                __builtin_prefetch(A + (size_t)(mBase + ldRow) * KDIM + k0 + 32 + ldHalf * 16, 0, 3);
                __builtin_prefetch(B + (size_t)(nBase + ldRow) * KDIM + k0 + 32 + ldHalf * 16, 0, 3);
            }
        }
        __syncthreads();

        // A fragments: lane<16 holds K 0-7 & 16-23, lane>=16 holds K 8-15 & 24-31
        v16h afr[2];
        #pragma unroll
        for (int tm = 0; tm < 2; ++tm) {
            const _Float16* base = sA + (wm * 32 + tm * 16 + lrow) * LDS_STRIDE_H + hsel * 8;
            const h8 lo = *reinterpret_cast<const h8*>(base);
            const h8 hi = *reinterpret_cast<const h8*>(base + 16);
            afr[tm] = __builtin_shufflevector(lo, hi, 0,1,2,3,4,5,6,7,8,9,10,11,12,13,14,15);
        }
        // B fragments: lane owns column N=(lane&15); K 0-7 & 16-23 (lane<16) or 8-15 & 24-31
        v16h bfr[4];
        #pragma unroll
        for (int tn = 0; tn < 4; ++tn) {
            const _Float16* base = sB + (wn * 64 + tn * 16 + lrow) * LDS_STRIDE_H + hsel * 8;
            const h8 lo = *reinterpret_cast<const h8*>(base);
            const h8 hi = *reinterpret_cast<const h8*>(base + 16);
            bfr[tn] = __builtin_shufflevector(lo, hi, 0,1,2,3,4,5,6,7,8,9,10,11,12,13,14,15);
        }
        #pragma unroll
        for (int tm = 0; tm < 2; ++tm)
            #pragma unroll
            for (int tn = 0; tn < 4; ++tn)
                acc[tm][tn] = __builtin_amdgcn_wmma_f32_16x16x32_f16(
                    false, afr[tm], false, bfr[tn], (short)0, acc[tm][tn], false, false);
    }

    // Store: VGPR r -> (M = r + 8*(lane>=16), N = lane&15) within each tile
    #pragma unroll
    for (int tm = 0; tm < 2; ++tm) {
        const int rbase = mBase + wm * 32 + tm * 16 + hsel * 8;
        #pragma unroll
        for (int tn = 0; tn < 4; ++tn) {
            const int col = nBase + wn * 64 + tn * 16 + lrow;
            #pragma unroll
            for (int r = 0; r < 8; ++r)
                C[(size_t)(rbase + r) * NPAD + col] = acc[tm][tn][r];
        }
    }
}

// ------------------------- epilogue kernels ---------------------------------
// L[row] = logsumexp_k(logits[row,k]/T) using the exact fixed shift 10
// (|logits| <= 1 since both operands are unit vectors).
__global__ void row_lse(const float* __restrict__ logits, float* __restrict__ L) {
    const int row = blockIdx.x, t = threadIdx.x;
    const float* p = logits + (size_t)row * NPAD;
    float s = 0.0f;
    for (int k = t; k < NPROTO; k += 256) s += __expf(p[k] * INV_T - 10.0f);
    __shared__ float sbuf[8];
    const float tot = blockReduceSum256(s, sbuf);
    if (t == 0) L[row] = 10.0f + __logf(tot);
}

// Q0[c][n,k] = exp(logits/eps - 20) (fixed shift; cancels in normalization).
// Accumulates global sum S[c].
__global__ void q0_kernel(const float* __restrict__ logits,
                          float* __restrict__ Q0, float* __restrict__ S) {
    const int n = blockIdx.x, c = blockIdx.y, t = threadIdx.x;
    const float* p = logits + (size_t)(c * BS + n) * NPAD;
    float* q = Q0 + (size_t)c * BS * NPROTO + (size_t)n * NPROTO;
    float s = 0.0f;
    for (int k = t; k < NPROTO; k += 256) {
        const float e = __expf(p[k] * INV_EPS - 20.0f);
        q[k] = e; s += e;
    }
    __shared__ float sbuf[8];
    const float tot = blockReduceSum256(s, sbuf);
    if (t == 0) atomicAdd(&S[c], tot);
}

__global__ void init_cvec(const float* __restrict__ S, float* __restrict__ cvec) {
    const int idx = blockIdx.x * 256 + threadIdx.x;
    if (idx < 2 * BS) cvec[idx] = 1.0f / S[idx >> 10];
}

// Sinkhorn row step (over prototypes k): rvec[k] = (1/K) / sum_n Q0[n,k]*cvec[n]
__global__ void sk_row(const float* __restrict__ Q0, const float* __restrict__ cvec,
                       float* __restrict__ rvec) {
    const int c = blockIdx.y;
    const int k = blockIdx.x * 256 + threadIdx.x;
    if (k >= NPROTO) return;
    const float* q  = Q0 + (size_t)c * BS * NPROTO;
    const float* cv = cvec + c * BS;
    float s = 0.0f;
    for (int n = 0; n < BS; ++n) s += q[(size_t)n * NPROTO + k] * cv[n];
    rvec[c * NPROTO + k] = (1.0f / (float)NPROTO) / s;
}

// Sinkhorn col step (over samples n): d[n] = sum_k Q0[n,k]*rvec[k];
// cvec[n] = (1/N)/d ; dvec keeps d (final column-normalizer).
__global__ void sk_col(const float* __restrict__ Q0, const float* __restrict__ rvec,
                       float* __restrict__ cvec, float* __restrict__ dvec) {
    const int n = blockIdx.x, c = blockIdx.y, t = threadIdx.x;
    const float* q  = Q0 + (size_t)c * BS * NPROTO + (size_t)n * NPROTO;
    const float* rv = rvec + c * NPROTO;
    float s = 0.0f;
    for (int k = t; k < NPROTO; k += 256) s += q[k] * rv[k];
    __shared__ float sbuf[8];
    const float tot = blockReduceSum256(s, sbuf);
    if (t == 0) { dvec[c * BS + n] = tot; cvec[c * BS + n] = (1.0f / (float)BS) / tot; }
}

// loss += (1/(2*7*1024)) * sum_{c,v!=c,n} ( L[v,n] - (1/T) * <q_c[n,:], logits[v,n,:]> )
// with q_c[n,k] = Q0[n,k]*rvec[k]/dvec[n] (row-stochastic, so sum_k q = 1).
__global__ void loss_kernel(const float* __restrict__ logits, const float* __restrict__ Q0,
                            const float* __restrict__ rvec, const float* __restrict__ dvec,
                            const float* __restrict__ L, float* __restrict__ out) {
    const int n = blockIdx.x, j = blockIdx.y, t = threadIdx.x;
    const int c  = j / 7;
    const int vv = j % 7;
    const int v  = (vv < c) ? vv : vv + 1;
    const int row = v * BS + n;
    const float* q  = Q0 + (size_t)c * BS * NPROTO + (size_t)n * NPROTO;
    const float* rv = rvec + c * NPROTO;
    const float* p  = logits + (size_t)row * NPAD;
    float s = 0.0f;
    for (int k = t; k < NPROTO; k += 256) s += q[k] * rv[k] * p[k];
    __shared__ float sbuf[8];
    const float dot = blockReduceSum256(s, sbuf);
    if (t == 0) {
        const float term = L[row] - (dot / dvec[c * BS + n]) * INV_T;
        atomicAdd(out, term * (1.0f / (2.0f * 7.0f * (float)BS)));
    }
}

// ------------------------- launcher -----------------------------------------
extern "C" void kernel_launch(void* const* d_in, const int* in_sizes, int n_in,
                              void* d_out, int out_size, void* d_ws, size_t ws_size,
                              hipStream_t stream) {
    const float* x = (const float*)d_in[0];   // 8192 x 256
    const float* W = (const float*)d_in[1];   // 3000 x 256
    float* out = (float*)d_out;

    char* ws = (char*)d_ws;
    size_t off = 0;
    auto carve = [&](size_t bytes) { char* p = ws + off; off = (off + bytes + 255) & ~(size_t)255; return p; };

    _Float16* z      = (_Float16*)carve((size_t)ROWS * KDIM * 2);      // 4 MB
    _Float16* Wn     = (_Float16*)carve((size_t)NPAD * KDIM * 2);      // 1.5 MB
    float*    logits = (float*)   carve((size_t)ROWS * NPAD * 4);      // 100.7 MB
    float*    L      = (float*)   carve((size_t)ROWS * 4);
    float*    Q0     = (float*)   carve((size_t)2 * BS * NPROTO * 4);  // 24.6 MB
    float*    S      = (float*)   carve(2 * 4);
    float*    rvec   = (float*)   carve((size_t)2 * NPROTO * 4);
    float*    cvec   = (float*)   carve((size_t)2 * BS * 4);
    float*    dvec   = (float*)   carve((size_t)2 * BS * 4);
    (void)ws_size; (void)in_sizes; (void)n_in; (void)out_size;

    zero_scalars<<<1, 32, 0, stream>>>(out, S);
    rownorm_f16<<<ROWS, 256, 0, stream>>>(x, z, ROWS);
    rownorm_f16<<<NPAD, 256, 0, stream>>>(W, Wn, NPROTO);   // rows >= 3000 zeroed

    dim3 gGemm(NPAD / 128, ROWS / 128);                      // (24, 64)
    gemm_f16_wmma<<<gGemm, 256, 0, stream>>>(z, Wn, logits);

    row_lse<<<ROWS, 256, 0, stream>>>(logits, L);
    q0_kernel<<<dim3(BS, 2), 256, 0, stream>>>(logits, Q0, S);
    init_cvec<<<(2 * BS + 255) / 256, 256, 0, stream>>>(S, cvec);

    for (int it = 0; it < 3; ++it) {
        sk_row<<<dim3((NPROTO + 255) / 256, 2), 256, 0, stream>>>(Q0, cvec, rvec);
        sk_col<<<dim3(BS, 2), 256, 0, stream>>>(Q0, rvec, cvec, dvec);
    }
    loss_kernel<<<dim3(BS, 14), 256, 0, stream>>>(logits, Q0, rvec, dvec, L, out);
}
